// SpikeNet_32598801776734
// MI455X (gfx1250) — matrix operand order repaired and checked by reading, not compiled, not run on hardware
//
#include <hip/hip_runtime.h>
#include <hip/hip_bf16.h>

// ---------------- constants from the reference ----------------
constexpr int kT      = 16;
constexpr int kNnodes = 50000;
constexpr int kInF    = 128;
constexpr int kB      = 4096;
constexpr int kS1     = 5;
constexpr int kS2     = 2;
constexpr int kH1     = 128;
constexpr int kH2     = 64;
constexpr int kC      = 32;
constexpr int kK      = 5;
constexpr int kRows1  = kB + kB * kS1;       // 24576 layer-1 rows
constexpr int kKdim   = 2 * kInF;            // 256 (self | neigh concat)

typedef __attribute__((ext_vector_type(16))) __bf16 v16bf;
typedef __attribute__((ext_vector_type(8)))  float  v8f;
typedef int v4i __attribute__((vector_size(16)));

#define GLOBAL_AS __attribute__((address_space(1)))
#define LDS_AS    __attribute__((address_space(3)))

// Async global->LDS DMA (ASYNCcnt-tracked) when the toolchain declares it.
#if defined(__HIP_DEVICE_COMPILE__) && \
    __has_builtin(__builtin_amdgcn_global_load_async_to_lds_b128)
#define HAVE_ASYNC_LDS 1
#else
#define HAVE_ASYNC_LDS 0
#endif

__device__ __forceinline__ void wait_async_zero() {
#if defined(__HIP_DEVICE_COMPILE__)
#if __has_builtin(__builtin_amdgcn_s_wait_asynccnt)
  __builtin_amdgcn_s_wait_asynccnt(0);
#else
  asm volatile("s_wait_asynccnt 0" ::: "memory");
#endif
#endif
}

// ---------------- weight swizzle into B-fragment order ----------------
// Logical W: K in [0,256): rows 0..127 from Ws (KxN row-major), 128..255 Wn.
// Output order [kt][nt][lane][e] so a lane's 16-elem B fragment is one
// contiguous 32-byte chunk. B-frag map used here (K order matches A loads): n = lane&15,
// K = (lane>>4)*16 + e.
__global__ void build_wfrag(const float* __restrict__ Ws,
                            const float* __restrict__ Wn,
                            __bf16* __restrict__ Wfrag, int N) {
  int idx = blockIdx.x * blockDim.x + threadIdx.x;
  int NT = N / 16;
  int total = 8 * NT * 32 * 16;            // 8 k-tiles of 32
  if (idx >= total) return;
  int e    = idx & 15;
  int lane = (idx >> 4) & 31;
  int tile = idx >> 9;                     // kt*NT + nt
  int kt = tile / NT, nt = tile % NT;
  int n = nt * 16 + (lane & 15);
  int K = kt * 32 + (lane >> 4) * 16 + e;
  float v = (K < kInF) ? Ws[K * N + n] : Wn[(K - kInF) * N + n];
  Wfrag[idx] = (__bf16)v;
}

// ---------------- per-timestep gather: build A1 (24576 x 256 bf16) -------
__global__ void gather_A1(const float* __restrict__ x,
                          const int* __restrict__ nodes,
                          const int* __restrict__ nbr1,
                          const int* __restrict__ nbr2,
                          __bf16* __restrict__ A1, int t) {
  int r = blockIdx.x;         // 0 .. kRows1-1
  int c = threadIdx.x;        // 0 .. 127
  const float* xt = x + (long long)t * kNnodes * kInF;
  const int* n1 = nbr1 + (long long)t * kB * kS1;
  const int* n2 = nbr2 + (long long)t * kB * kS1 * kS2;
  int self_node;
  float neigh;
  if (r < kB) {
    self_node = nodes[r];
    float s = 0.f;
#pragma unroll
    for (int j = 0; j < kS1; ++j)
      s += xt[(long long)n1[r * kS1 + j] * kInF + c];
    neigh = s * (1.0f / kS1);
  } else {
    int j = r - kB;
    self_node = n1[j];
    float s = 0.f;
#pragma unroll
    for (int q = 0; q < kS2; ++q)
      s += xt[(long long)n2[j * kS2 + q] * kInF + c];
    neigh = s * (1.0f / kS2);
  }
  float sv = xt[(long long)self_node * kInF + c];
  __bf16* arow = A1 + (long long)r * kKdim;
  arow[c]        = (__bf16)sv;
  arow[kInF + c] = (__bf16)neigh;
}

// ---- stage fragment-ordered weights (ELEMS bf16) into LDS, 256 threads --
template <int ELEMS>
__device__ __forceinline__ void stage_weights(const __bf16* __restrict__ Wfrag,
                                              __bf16* sW, int tid) {
  constexpr int CHUNKS = ELEMS / 8;        // 16-byte chunks
#if HAVE_ASYNC_LDS
#pragma unroll
  for (int i = 0; i < CHUNKS / 256; ++i) {
    int ofs = (tid + i * 256) * 8;
    __builtin_amdgcn_global_load_async_to_lds_b128(
        (GLOBAL_AS v4i*)(Wfrag + ofs), (LDS_AS v4i*)(sW + ofs), 0, 0);
  }
  wait_async_zero();
#else
#pragma unroll
  for (int i = 0; i < CHUNKS / 256; ++i) {
    int ofs = (tid + i * 256) * 8;
    *(uint4*)(sW + ofs) = *(const uint4*)(Wfrag + ofs);
  }
#endif
  __syncthreads();
}

// ---------------- WMMA layer 1: (24576x256)@(256x128) + b1, spike --------
// 8 waves/block; each wave owns one 16-row strip across all 8 N-tiles.
__global__ __launch_bounds__(256)
void wmma_l1(const __bf16* __restrict__ A1, const __bf16* __restrict__ Wfrag,
             const float* __restrict__ b1, __bf16* __restrict__ S1buf) {
  __shared__ __align__(16) __bf16 sW[8 * 8 * 32 * 16];   // 64 KB
  const int lane = threadIdx.x;     // 0..31
  const int wave = threadIdx.y;     // 0..7
  stage_weights<8 * 8 * 32 * 16>(Wfrag, sW, wave * 32 + lane);

  const int rt   = blockIdx.x * 8 + wave;        // 16-row tile index
  const int m    = lane & 15, half = lane >> 4;
  const __bf16* arow = A1 + (long long)(rt * 16 + m) * kKdim;

  v8f acc[8] = {};
#pragma unroll
  for (int kt = 0; kt < 8; ++kt) {
    int k0 = kt * 32;
    union { v16bf v; uint4 q[2]; } a;
    // A frag (this code's K order, matching B): per lane, 16 consecutive K at k0+16*half.
    a.q[0] = *(const uint4*)(arow + k0 + half * 16);
    a.q[1] = *(const uint4*)(arow + k0 + half * 16 + 8);
#pragma unroll
    for (int nt = 0; nt < 8; ++nt) {
      union { v16bf v; uint4 q[2]; } b;
      const __bf16* bp = sW + (((kt * 8 + nt) * 32 + lane) << 4);
      b.q[0] = *(const uint4*)(bp);        // ds_load_b128
      b.q[1] = *(const uint4*)(bp + 8);
      acc[nt] = __builtin_amdgcn_wmma_f32_16x16x32_bf16(
          false, a.v, false, b.v, (short)0, acc[nt], false, false);
    }
  }
  const int mbase = rt * 16 + half * 8;          // C/D layout: M = 8*half + r
#pragma unroll
  for (int nt = 0; nt < 8; ++nt) {
    int n = nt * 16 + (lane & 15);
    float bias = b1[n];
#pragma unroll
    for (int r = 0; r < 8; ++r) {
      float u = acc[nt][r] + bias;               // v1 = u1 since TAU==1
      S1buf[(long long)(mbase + r) * kH1 + n] =
          (__bf16)((u >= 1.0f) ? 1.0f : 0.0f);
    }
  }
}

// ---------------- build A2 = [g0 | mean5(g1)] (4096 x 256 bf16) ----------
__global__ void build_A2(const __bf16* __restrict__ S1buf,
                         __bf16* __restrict__ A2) {
  int b = blockIdx.x;      // 0..4095
  int h = threadIdx.x;     // 0..127
  A2[(long long)b * kKdim + h] = S1buf[(long long)b * kH1 + h];
  float s = 0.f;
#pragma unroll
  for (int j = 0; j < kS1; ++j)
    s += (float)S1buf[(long long)(kB + b * kS1 + j) * kH1 + h];
  A2[(long long)b * kKdim + kInF + h] = (__bf16)(s * (1.0f / kS1));
}

// ---------------- WMMA layer 2: (4096x256)@(256x64) + b2, spike ----------
__global__ __launch_bounds__(256)
void wmma_l2(const __bf16* __restrict__ A2, const __bf16* __restrict__ Wfrag,
             const float* __restrict__ b2, float* __restrict__ SPK, int t) {
  __shared__ __align__(16) __bf16 sW[8 * 4 * 32 * 16];   // 32 KB
  const int lane = threadIdx.x;
  const int wave = threadIdx.y;
  stage_weights<8 * 4 * 32 * 16>(Wfrag, sW, wave * 32 + lane);

  const int rt   = blockIdx.x * 8 + wave;
  const int m    = lane & 15, half = lane >> 4;
  const __bf16* arow = A2 + (long long)(rt * 16 + m) * kKdim;

  v8f acc[4] = {};
#pragma unroll
  for (int kt = 0; kt < 8; ++kt) {
    int k0 = kt * 32;
    union { v16bf v; uint4 q[2]; } a;
    a.q[0] = *(const uint4*)(arow + k0 + half * 16);
    a.q[1] = *(const uint4*)(arow + k0 + half * 16 + 8);
#pragma unroll
    for (int nt = 0; nt < 4; ++nt) {
      union { v16bf v; uint4 q[2]; } b;
      const __bf16* bp = sW + (((kt * 4 + nt) * 32 + lane) << 4);
      b.q[0] = *(const uint4*)(bp);
      b.q[1] = *(const uint4*)(bp + 8);
      acc[nt] = __builtin_amdgcn_wmma_f32_16x16x32_bf16(
          false, a.v, false, b.v, (short)0, acc[nt], false, false);
    }
  }
  const int mbase = rt * 16 + half * 8;
#pragma unroll
  for (int nt = 0; nt < 4; ++nt) {
    int n = nt * 16 + (lane & 15);
    float bias = b2[n];
#pragma unroll
    for (int r = 0; r < 8; ++r) {
      float u = acc[nt][r] + bias;
      int bb = mbase + r;
      SPK[((long long)bb * kT + t) * kH2 + n] = (u >= 1.0f) ? 1.0f : 0.0f;
    }
  }
}

// ---------------- delay mix + depthwise conv + readout -------------------
__global__ __launch_bounds__(64)
void readout(const float* __restrict__ SPK, const float* __restrict__ delay_w,
             const float* __restrict__ dwk, const float* __restrict__ roW,
             const float* __restrict__ rob, float* __restrict__ out) {
  int b = blockIdx.x;     // 0..4095
  int h = threadIdx.x;    // 0..63
  int g = h >> 3;         // group (gs = H2/GROUPS = 8)
  float w0 = delay_w[g * 3 + 0], w1 = delay_w[g * 3 + 1], w2 = delay_w[g * 3 + 2];
  float mx = fmaxf(w0, fmaxf(w1, w2));
  float e0 = __expf(w0 - mx), e1 = __expf(w1 - mx), e2 = __expf(w2 - mx);
  float inv = 1.0f / (e0 + e1 + e2);
  e0 *= inv; e1 *= inv; e2 *= inv;
  const float* sp = SPK + (long long)b * kT * kH2 + h;   // stride kH2 over t
  float dly[kT];
#pragma unroll
  for (int t = 0; t < kT; ++t) {
    float a = 0.f;
    if (t >= 1) a += e0 * sp[(t - 1) * kH2];
    if (t >= 3) a += e1 * sp[(t - 3) * kH2];
    if (t >= 5) a += e2 * sp[(t - 5) * kH2];
    dly[t] = a;
  }
  // depthwise conv (cross-correlation), K=5, SAME; then mean over T
  float msum = 0.f;
#pragma unroll
  for (int t = 0; t < kT; ++t) {
    float y = 0.f;
#pragma unroll
    for (int k = 0; k < kK; ++k) {
      int tt = t + k - 2;
      if (tt >= 0 && tt < kT) y += dwk[h * kK + k] * dly[tt];
    }
    msum += y;
  }
  msum *= (1.0f / kT);
  __shared__ float sm[kH2];
  sm[h] = msum;
  __syncthreads();
  if (h < kC) {
    float o = rob[h];
#pragma unroll
    for (int j = 0; j < kH2; ++j) o += sm[j] * roW[j * kC + h];
    out[(long long)b * kC + h] = o;
  }
}

// ---------------- host orchestration -------------------------------------
extern "C" void kernel_launch(void* const* d_in, const int* in_sizes, int n_in,
                              void* d_out, int out_size, void* d_ws, size_t ws_size,
                              hipStream_t stream) {
  const float* x       = (const float*)d_in[0];
  const int*   nodes   = (const int*)d_in[1];
  const int*   nbr1    = (const int*)d_in[2];
  const int*   nbr2    = (const int*)d_in[3];
  const float* W1s     = (const float*)d_in[4];
  const float* W1n     = (const float*)d_in[5];
  const float* b1      = (const float*)d_in[6];
  const float* W2s     = (const float*)d_in[7];
  const float* W2n     = (const float*)d_in[8];
  const float* b2      = (const float*)d_in[9];
  const float* delay_w = (const float*)d_in[10];
  const float* dwk     = (const float*)d_in[11];
  const float* roW     = (const float*)d_in[12];
  const float* rob     = (const float*)d_in[13];
  float* out = (float*)d_out;

  // bump allocator over workspace (all buffers fully written before read)
  char* ws = (char*)d_ws;
  size_t off = 0;
  auto alloc = [&](size_t bytes) {
    void* p = ws + off;
    off = (off + bytes + 255) & ~(size_t)255;
    return p;
  };
  __bf16* A1    = (__bf16*)alloc((size_t)kRows1 * kKdim * 2);   // 12.6 MB
  __bf16* S1buf = (__bf16*)alloc((size_t)kRows1 * kH1 * 2);     //  6.3 MB
  __bf16* A2    = (__bf16*)alloc((size_t)kB * kKdim * 2);       //  2.1 MB
  float*  SPK   = (float*) alloc((size_t)kB * kT * kH2 * 4);    // 16.8 MB
  __bf16* W1f   = (__bf16*)alloc((size_t)8 * 8 * 32 * 16 * 2);  //  64 KB
  __bf16* W2f   = (__bf16*)alloc((size_t)8 * 4 * 32 * 16 * 2);  //  32 KB
  if (off > ws_size) return;   // workspace too small (deterministic no-op)

  build_wfrag<<<(8 * 8 * 512 + 255) / 256, 256, 0, stream>>>(W1s, W1n, W1f, kH1);
  build_wfrag<<<(8 * 4 * 512 + 255) / 256, 256, 0, stream>>>(W2s, W2n, W2f, kH2);

  for (int t = 0; t < kT; ++t) {
    gather_A1<<<kRows1, kInF, 0, stream>>>(x, nodes, nbr1, nbr2, A1, t);
    wmma_l1<<<kRows1 / 128, dim3(32, 8), 0, stream>>>(A1, W1f, b1, S1buf);
    build_A2<<<kB, kInF, 0, stream>>>(S1buf, A2);
    wmma_l2<<<kB / 128, dim3(32, 8), 0, stream>>>(A2, W2f, b2, SPK, t);
  }

  readout<<<kB, kH2, 0, stream>>>(SPK, delay_w, dwk, roW, rob, out);
}
